// TemporalFocus_53807350284292
// MI455X (gfx1250) — matrix operand order repaired and checked by reading, not compiled, not run on hardware
//
#include <hip/hip_runtime.h>
#include <stdint.h>

typedef __attribute__((ext_vector_type(16))) __bf16 v16bf;
typedef __attribute__((ext_vector_type(8)))  float  v8f;
typedef __attribute__((ext_vector_type(4)))  int    v4i;

#define AS1 __attribute__((address_space(1)))
#define AS3 __attribute__((address_space(3)))

#define B_  4
#define T_  2048
#define C_  1024
#define H_  16
#define HD_ 64

__device__ __forceinline__ unsigned short f2bf(float f) {
  unsigned u = __builtin_bit_cast(unsigned, f);
  unsigned r = (u + 0x7FFFu + ((u >> 16) & 1u)) >> 16;  // RNE
  return (unsigned short)r;
}

union FragB {
  unsigned u[8];
  uint4    q[2];
  v16bf    v;
};

__device__ __forceinline__ v8f v8f_zero() {
  v8f z;
  #pragma unroll
  for (int i = 0; i < 8; i++) z[i] = 0.0f;
  return z;
}

__device__ __forceinline__ v8f wmma_bf16(v16bf a, v16bf b, v8f c) {
  // D = A(16x32 bf16) * B(32x16 bf16) + C(16x16 f32)
  return __builtin_amdgcn_wmma_f32_16x16x32_bf16(false, a, false, b, (short)0, c,
                                                 false, false);
}

// Async global->LDS 16-byte copy (ASYNCcnt-tracked DMA, no VGPR round-trip).
__device__ __forceinline__ void async_b128(void* lds, const void* g) {
#if __has_builtin(__builtin_amdgcn_global_load_async_to_lds_b128)
  __builtin_amdgcn_global_load_async_to_lds_b128(
      (AS1 v4i*)g, (AS3 v4i*)lds, 0, 0);
#else
  unsigned laddr = (unsigned)(unsigned long)(AS3 void*)lds;
  asm volatile("global_load_async_to_lds_b128 %0, %1, off"
               :: "v"(laddr), "v"(g) : "memory");
#endif
}

__device__ __forceinline__ void wait_async0() {
#if __has_builtin(__builtin_amdgcn_s_wait_asynccnt)
  __builtin_amdgcn_s_wait_asynccnt(0);
#else
  asm volatile("s_wait_asynccnt 0x0" ::: "memory");
#endif
}

// ---------------------------------------------------------------- convert
__global__ void k_cvt(const float* __restrict__ src,
                      unsigned short* __restrict__ dst, int n) {
  int i = blockIdx.x * blockDim.x + threadIdx.x;
  int stride = gridDim.x * blockDim.x;
  for (; i < n; i += stride) dst[i] = f2bf(src[i]);
}

// ---------------------------------------------------------------- GEMM
// C[M,N] = A[M,K](bf16) * B[K,N](bf16) + bias[N]
// MODE 0: scatter bf16 into q/k/v [B,H,T,hd].  MODE 1: fp32 to `of`.
template <int MODE>
__global__ __launch_bounds__(256) void k_gemm(
    const unsigned short* __restrict__ A, const unsigned short* __restrict__ Bw,
    const float* __restrict__ bias, int M, int N, int K,
    unsigned short* __restrict__ oq, unsigned short* __restrict__ ok,
    unsigned short* __restrict__ ov, float* __restrict__ of) {
  __shared__ __align__(16) unsigned short sA[128][32];   // A tile  [m][k]
  __shared__ __align__(16) unsigned short sBt[128][32];  // B tile  [n][k] (transposed)

  const int tid  = threadIdx.x;
  const int lane = tid & 31, wid = tid >> 5;
  const int half = lane >> 4, lm = lane & 15;
  const int wm = wid >> 2, wn = wid & 3;         // wave tile: 64m x 32n
  const int bm = blockIdx.x * 128, bn = blockIdx.y * 128;

  v8f acc[4][2];
  #pragma unroll
  for (int i = 0; i < 4; i++)
    #pragma unroll
    for (int j = 0; j < 2; j++) acc[i][j] = v8f_zero();

  const int arow = tid >> 1, acol = (tid & 1) * 16;   // 16 elems / thread
  const int bkrow = tid >> 3, bn0 = (tid & 7) * 16;   // 16 elems / thread

  for (int k0 = 0; k0 < K; k0 += 32) {
    {  // A tile: straight copy -> async DMA to LDS
      const unsigned short* g = A + (size_t)(bm + arow) * K + k0 + acol;
      async_b128(&sA[arow][acol],     g);
      async_b128(&sA[arow][acol + 8], g + 8);
    }
    {  // B tile: coalesced load, transposed LDS store (async can't transpose)
      const unsigned short* g = Bw + (size_t)(k0 + bkrow) * N + bn + bn0;
      uint4 v0 = *(const uint4*)g;
      uint4 v1 = *(const uint4*)(g + 8);
      const unsigned short* e0 = (const unsigned short*)&v0;
      const unsigned short* e1 = (const unsigned short*)&v1;
      #pragma unroll
      for (int e = 0; e < 8; e++) sBt[bn0 + e][bkrow] = e0[e];
      #pragma unroll
      for (int e = 0; e < 8; e++) sBt[bn0 + 8 + e][bkrow] = e1[e];
    }
    wait_async0();
    __syncthreads();

    #pragma unroll
    for (int ti = 0; ti < 4; ti++) {
      FragB af;  // A-frag: lane holds row (wm*64+ti*16+lm), K chunks per ISA layout
      const int r = wm * 64 + ti * 16 + lm;
      af.q[0] = *(const uint4*)&sA[r][half * 8];
      af.q[1] = *(const uint4*)&sA[r][16 + half * 8];
      #pragma unroll
      for (int tj = 0; tj < 2; tj++) {
        FragB bf;  // B-frag: lane holds col n, K chunks per ISA layout (half*8, 16+half*8)
        const int c = wn * 32 + tj * 16 + lm;
        bf.q[0] = *(const uint4*)&sBt[c][half * 8];
        bf.q[1] = *(const uint4*)&sBt[c][16 + half * 8];
        acc[ti][tj] = wmma_bf16(af.v, bf.v, acc[ti][tj]);
      }
    }
    __syncthreads();
  }

  // epilogue: C layout row = i + 8*half, col = lane&15
  #pragma unroll
  for (int ti = 0; ti < 4; ti++)
    #pragma unroll
    for (int tj = 0; tj < 2; tj++)
      #pragma unroll
      for (int i = 0; i < 8; i++) {
        const int row = bm + wm * 64 + ti * 16 + half * 8 + i;
        const int col = bn + wn * 32 + tj * 16 + lm;
        const float val = acc[ti][tj][i] + bias[col];
        if (MODE == 0) {
          const int which = col >> 10, cc = col & 1023;
          const int h = cc >> 6, d = cc & 63;
          const int bb = row >> 11, t = row & 2047;
          const size_t idx = (((size_t)bb * H_ + h) * T_ + t) * HD_ + d;
          const unsigned short bv = f2bf(val);
          if (which == 0)      oq[idx] = bv;
          else if (which == 1) ok[idx] = bv;
          else                 ov[idx] = bv;
        } else {
          of[(size_t)row * N + col] = val;
        }
      }
}

// ---------------------------------------------------------------- flash attention
// grid (T/64, B*H), 128 threads (4 waves). Each wave owns 16 query rows.
__global__ __launch_bounds__(128) void k_attn(
    const unsigned short* __restrict__ q, const unsigned short* __restrict__ k,
    const unsigned short* __restrict__ v, unsigned short* __restrict__ y) {
  __shared__ __align__(16) unsigned short sK[64][64];       // [key][d]
  __shared__ __align__(16) unsigned short sVt[64][64];      // [d][key]
  __shared__ __align__(16) unsigned short sP[4][16][64];    // per-wave P staging

  const int tid = threadIdx.x, lane = tid & 31, wid = tid >> 5;
  const int half = lane >> 4, lm = lane & 15;
  const int qbase = blockIdx.x * 64;
  const int bh = blockIdx.y;
  const int b = bh >> 4, h = bh & 15;
  const float scale = 0.125f;  // 1/sqrt(64)

  // Q fragments: 16 rows x 64 d, two 16x32 A-frags
  FragB qf[2];
  {
    const unsigned short* qp =
        q + ((size_t)bh * T_ + (qbase + wid * 16 + lm)) * HD_;
    #pragma unroll
    for (int f = 0; f < 2; f++) {
      qf[f].q[0] = *(const uint4*)(qp + f * 32 + half * 8);
      qf[f].q[1] = *(const uint4*)(qp + f * 32 + 16 + half * 8);
    }
  }

  v8f o[4];
  #pragma unroll
  for (int i = 0; i < 4; i++) o[i] = v8f_zero();
  float mrow[8], lrow[8];
  #pragma unroll
  for (int i = 0; i < 8; i++) { mrow[i] = -3.0e38f; lrow[i] = 0.0f; }

  const int kend = qbase + 64;                 // block-uniform causal bound
  const int srow = tid >> 1, scol = (tid & 1) * 32;  // 32 elems / thread staging

  for (int kk0 = 0; kk0 < kend; kk0 += 64) {
    {  // K tile straight copy -> async DMA to LDS
      const unsigned short* g = k + ((size_t)bh * T_ + kk0 + srow) * HD_ + scol;
      #pragma unroll
      for (int c = 0; c < 4; c++)
        async_b128(&sK[srow][scol + c * 8], g + c * 8);
    }
    {  // V tile transposed (manual)
      const unsigned short* g = v + ((size_t)bh * T_ + kk0 + srow) * HD_ + scol;
      #pragma unroll
      for (int c = 0; c < 4; c++) {
        uint4 vv = *(const uint4*)(g + c * 8);
        const unsigned short* e = (const unsigned short*)&vv;
        #pragma unroll
        for (int j = 0; j < 8; j++) sVt[scol + c * 8 + j][srow] = e[j];
      }
    }
    wait_async0();
    __syncthreads();

    // S = Q K^T : 4 tiles of 16 keys, 2 WMMAs each (d = 0..31, 32..63)
    v8f s[4];
    #pragma unroll
    for (int j = 0; j < 4; j++) {
      FragB kf;
      kf.q[0] = *(const uint4*)&sK[j * 16 + lm][half * 8];
      kf.q[1] = *(const uint4*)&sK[j * 16 + lm][16 + half * 8];
      s[j] = wmma_bf16(qf[0].v, kf.v, v8f_zero());
      kf.q[0] = *(const uint4*)&sK[j * 16 + lm][32 + half * 8];
      kf.q[1] = *(const uint4*)&sK[j * 16 + lm][32 + 16 + half * 8];
      s[j] = wmma_bf16(qf[1].v, kf.v, s[j]);
    }

    // scale + causal mask + row max
    float tmax[8];
    #pragma unroll
    for (int i = 0; i < 8; i++) tmax[i] = -3.0e38f;
    const int rowg0 = qbase + wid * 16 + half * 8;
    #pragma unroll
    for (int j = 0; j < 4; j++) {
      const int keyg = kk0 + j * 16 + lm;
      #pragma unroll
      for (int i = 0; i < 8; i++) {
        float sv = s[j][i] * scale;
        if (keyg > rowg0 + i) sv = -3.0e38f;
        s[j][i] = sv;
        tmax[i] = fmaxf(tmax[i], sv);
      }
    }
    #pragma unroll
    for (int i = 0; i < 8; i++) {  // row reduce across the 16 col-lanes
      float t = tmax[i];
      t = fmaxf(t, __shfl_xor(t, 1, 32));
      t = fmaxf(t, __shfl_xor(t, 2, 32));
      t = fmaxf(t, __shfl_xor(t, 4, 32));
      t = fmaxf(t, __shfl_xor(t, 8, 32));
      tmax[i] = t;
    }
    float alpha[8];
    #pragma unroll
    for (int i = 0; i < 8; i++) {
      const float nm = fmaxf(mrow[i], tmax[i]);
      alpha[i] = __expf(mrow[i] - nm);
      mrow[i] = nm;
    }
    float lsum[8];
    #pragma unroll
    for (int i = 0; i < 8; i++) lsum[i] = 0.0f;
    #pragma unroll
    for (int j = 0; j < 4; j++)
      #pragma unroll
      for (int i = 0; i < 8; i++) {
        const float p = __expf(s[j][i] - mrow[i]);
        s[j][i] = p;
        lsum[i] += p;
      }
    #pragma unroll
    for (int i = 0; i < 8; i++) {
      float t = lsum[i];
      t += __shfl_xor(t, 1, 32);
      t += __shfl_xor(t, 2, 32);
      t += __shfl_xor(t, 4, 32);
      t += __shfl_xor(t, 8, 32);
      lrow[i] = lrow[i] * alpha[i] + t;
    }
    #pragma unroll
    for (int dt = 0; dt < 4; dt++)
      #pragma unroll
      for (int i = 0; i < 8; i++) o[dt][i] *= alpha[i];

    // C-layout -> A-layout via per-wave LDS staging of P (bf16)
    #pragma unroll
    for (int j = 0; j < 4; j++)
      #pragma unroll
      for (int i = 0; i < 8; i++)
        sP[wid][half * 8 + i][j * 16 + lm] = f2bf(s[j][i]);

    FragB pf[2];
    #pragma unroll
    for (int f = 0; f < 2; f++) {
      pf[f].q[0] = *(const uint4*)&sP[wid][lm][f * 32 + half * 8];
      pf[f].q[1] = *(const uint4*)&sP[wid][lm][f * 32 + 16 + half * 8];
    }
    #pragma unroll
    for (int dt = 0; dt < 4; dt++) {
      FragB vf;
      vf.q[0] = *(const uint4*)&sVt[dt * 16 + lm][half * 8];
      vf.q[1] = *(const uint4*)&sVt[dt * 16 + lm][16 + half * 8];
      o[dt] = wmma_bf16(pf[0].v, vf.v, o[dt]);
      vf.q[0] = *(const uint4*)&sVt[dt * 16 + lm][32 + half * 8];
      vf.q[1] = *(const uint4*)&sVt[dt * 16 + lm][32 + 16 + half * 8];
      o[dt] = wmma_bf16(pf[1].v, vf.v, o[dt]);
    }
    __syncthreads();
  }

  // y[B,T,C] bf16, reassembled head-major columns
  #pragma unroll
  for (int dt = 0; dt < 4; dt++)
    #pragma unroll
    for (int i = 0; i < 8; i++) {
      const int rowg = qbase + wid * 16 + half * 8 + i;
      const int d = dt * 16 + lm;
      const float val = o[dt][i] / lrow[i];
      y[((size_t)(b * T_ + rowg)) * C_ + h * HD_ + d] = f2bf(val);
    }
}

// ---------------------------------------------------------------- launch
extern "C" void kernel_launch(void* const* d_in, const int* in_sizes, int n_in,
                              void* d_out, int out_size, void* d_ws, size_t ws_size,
                              hipStream_t stream) {
  (void)in_sizes; (void)n_in; (void)out_size; (void)ws_size;
  const float* x  = (const float*)d_in[0];
  const float* w1 = (const float*)d_in[1];
  const float* b1 = (const float*)d_in[2];
  const float* w2 = (const float*)d_in[3];
  const float* b2 = (const float*)d_in[4];
  float* out = (float*)d_out;

  unsigned short* ws = (unsigned short*)d_ws;
  const size_t nX  = (size_t)B_ * T_ * C_;       // 8388608
  const size_t nW1 = (size_t)C_ * 3 * C_;        // 3145728
  const size_t nW2 = (size_t)C_ * C_;            // 1048576
  const size_t nHT = (size_t)B_ * H_ * T_ * HD_; // 8388608
  unsigned short* xb  = ws;
  unsigned short* w1b = xb + nX;
  unsigned short* w2b = w1b + nW1;
  unsigned short* qws = w2b + nW2;
  unsigned short* kws = qws + nHT;
  unsigned short* vws = kws + nHT;
  unsigned short* yws = vws + nHT;

  k_cvt<<<4096, 256, 0, stream>>>(x,  xb,  (int)nX);
  k_cvt<<<4096, 256, 0, stream>>>(w1, w1b, (int)nW1);
  k_cvt<<<2048, 256, 0, stream>>>(w2, w2b, (int)nW2);

  dim3 blk(256);
  dim3 g1(64, 24);  // M/128 x 3C/128
  k_gemm<0><<<g1, blk, 0, stream>>>(xb, w1b, b1, B_ * T_, 3 * C_, C_,
                                    qws, kws, vws, nullptr);

  dim3 g2(T_ / 64, B_ * H_);
  k_attn<<<g2, dim3(128), 0, stream>>>(qws, kws, vws, yws);

  dim3 g3(64, 8);   // M/128 x C/128
  k_gemm<1><<<g3, blk, 0, stream>>>(yws, w2b, b2, B_ * T_, C_, C_,
                                    nullptr, nullptr, nullptr, out);
}
